// cline_fea_extract_64948495450573
// MI455X (gfx1250) — matrix operand order repaired and checked by reading, not compiled, not run on hardware
//
#include <hip/hip_runtime.h>
#include <stdint.h>

// ---------------------------------------------------------------------------
// GCN pipeline for MI455X (gfx1250, wave32).
// Bound: edge scatter/gather (~3.5 GB @ 23.3 TB/s ≈ 150 us). GEMMs (6.6 GFLOP)
// run as bf16 WMMA (v_wmma_f32_16x16x32_bf16, f32 accumulate).
// ---------------------------------------------------------------------------

typedef __bf16 bf16_t;
typedef __attribute__((ext_vector_type(16))) __bf16 v16bf;
typedef __attribute__((ext_vector_type(8)))  __bf16 v8bf;
typedef __attribute__((ext_vector_type(8)))  float  v8f;

#define BN_EPS 1e-5f

__device__ __forceinline__ bf16_t f2bf(float x) {
  // round-to-nearest-even f32 -> bf16
  unsigned int u = __float_as_uint(x);
  unsigned int r = (u + 0x7FFFu + ((u >> 16) & 1u)) >> 16;
  unsigned short h = (unsigned short)r;
  return __builtin_bit_cast(bf16_t, h);
}

// ------------------------------ gcn_norm ----------------------------------

__global__ void init_deg_kernel(float* __restrict__ deg, int n) {
  int i = blockIdx.x * blockDim.x + threadIdx.x;
  if (i < n) deg[i] = 1.0f;  // self-loop
}

__global__ void edge_deg_kernel(const int* __restrict__ row, float* __restrict__ deg, int e) {
  int i = blockIdx.x * blockDim.x + threadIdx.x;
  if (i < e) atomicAdd(&deg[row[i]], 1.0f);
}

__global__ void dinv_kernel(const float* __restrict__ deg, float* __restrict__ dinv, int n) {
  int i = blockIdx.x * blockDim.x + threadIdx.x;
  if (i < n) dinv[i] = rsqrtf(deg[i]);  // deg >= 1 always (self-loop)
}

__global__ void enorm_kernel(const int* __restrict__ row, const int* __restrict__ col,
                             const float* __restrict__ dinv, float* __restrict__ nrm, int e) {
  int i = blockIdx.x * blockDim.x + threadIdx.x;
  if (i < e) nrm[i] = dinv[row[i]] * dinv[col[i]];
}

// ------------------------------ BatchNorm ---------------------------------

__global__ void bn_zero_kernel(float* __restrict__ gsum, float* __restrict__ gsq, int f) {
  int i = threadIdx.x;
  if (i < f) { gsum[i] = 0.0f; gsq[i] = 0.0f; }
}

// blockDim = (32,8); grid = (F/32, CHUNKS). Coalesced 32-feature columns.
__global__ void bn_partial_kernel(const float* __restrict__ X, float* __restrict__ gsum,
                                  float* __restrict__ gsq, int n, int f) {
  __shared__ float sh_s[8][32];
  __shared__ float sh_q[8][32];
  const int tx = threadIdx.x, ty = threadIdx.y;
  const int fcol = blockIdx.x * 32 + tx;
  float s = 0.0f, q = 0.0f;
  for (int i = blockIdx.y * 8 + ty; i < n; i += gridDim.y * 8) {
    float v = X[(size_t)i * f + fcol];
    s += v; q += v * v;
  }
  sh_s[ty][tx] = s; sh_q[ty][tx] = q;
  __syncthreads();
  if (ty == 0) {
    for (int r = 1; r < 8; ++r) { s += sh_s[r][tx]; q += sh_q[r][tx]; }
    atomicAdd(&gsum[fcol], s);
    atomicAdd(&gsq[fcol], q);
  }
}

__global__ void bn_finalize_kernel(const float* __restrict__ gsum, const float* __restrict__ gsq,
                                   float* __restrict__ mu, float* __restrict__ rsig,
                                   int n, int f) {
  int i = threadIdx.x;
  if (i < f) {
    float m   = gsum[i] / (float)n;
    float var = gsq[i]  / (float)n - m * m;   // biased variance (matches jnp.var)
    mu[i]   = m;
    rsig[i] = rsqrtf(var + BN_EPS);
  }
}

__global__ void bn_apply_kernel(const float* __restrict__ X, const float* __restrict__ mu,
                                const float* __restrict__ rsig, const float* __restrict__ g,
                                const float* __restrict__ b, bf16_t* __restrict__ Xbf,
                                size_t total, int f) {
  size_t idx = (size_t)blockIdx.x * blockDim.x + threadIdx.x;
  if (idx >= total) return;
  int fc = (int)(idx & (size_t)(f - 1));  // f is a power of two (128/256)
  float v = (X[idx] - mu[fc]) * rsig[fc] * g[fc] + b[fc];
  Xbf[idx] = f2bf(v);
}

// --------------------------- weight transpose ------------------------------
// W [K,128] f32 row-major  ->  Wt [128,K] bf16 (column-major of W)
__global__ void wt_cvt_kernel(const float* __restrict__ W, bf16_t* __restrict__ Wt, int K) {
  int idx = blockIdx.x * blockDim.x + threadIdx.x;
  if (idx >= 128 * K) return;
  int n = idx / K, k = idx - n * K;
  Wt[idx] = f2bf(W[(size_t)k * 128 + n]);
}

// ------------------------------ WMMA GEMM ---------------------------------
// Y[N,128] = A[N,K](bf16) @ W[K,128], W pre-transposed as Wt[128,K](bf16).
// One wave per 16-row strip; 8 col tiles => 8 v_wmma per K-step of 32.
// Fragment layouts per CDNA5 ISA §7.12.2 (16-bit A 16x32, 16-bit B 32x16,
// 32-bit C/D 16x16). EXEC all-ones: N % 16 == 0, full 32-thread blocks.
__global__ void __launch_bounds__(32)
gemm_wmma_kernel(const bf16_t* __restrict__ A, const bf16_t* __restrict__ Wt,
                 float* __restrict__ Y, int K) {
  const int lane = (int)(threadIdx.x & 31u);
  const int half = lane >> 4;   // 0: lanes 0-15, 1: lanes 16-31
  const int sub  = lane & 15;
  const int rowBase = blockIdx.x * 16;

  v8f acc[8];
#pragma unroll
  for (int t = 0; t < 8; ++t)
#pragma unroll
    for (int j = 0; j < 8; ++j) acc[t][j] = 0.0f;

  const bf16_t* arow = A + (size_t)(rowBase + sub) * K;  // A row M = sub (both halves)

  for (int k = 0; k < K; k += 32) {
    // A 16x32 bf16 fragment: lane(half,sub) holds K = k+8*half+[0,8) and k+16+8*half+[0,8)
    v8bf a0 = *(const v8bf*)(arow + k + half * 8);
    v8bf a1 = *(const v8bf*)(arow + k + 16 + half * 8);
    v16bf a;
#pragma unroll
    for (int j = 0; j < 8; ++j) { a[j] = a0[j]; a[8 + j] = a1[j]; }

#pragma unroll
    for (int t = 0; t < 8; ++t) {
      // B 32x16 bf16 fragment: lane holds column N = t*16+sub, K = k+8*half+[0,8) and k+16+8*half+[0,8)
      const bf16_t* bcol = Wt + (size_t)(t * 16 + sub) * K + k + half * 8;
      v8bf b0 = *(const v8bf*)(bcol);
      v8bf b1 = *(const v8bf*)(bcol + 16);
      v16bf b;
#pragma unroll
      for (int j = 0; j < 8; ++j) { b[j] = b0[j]; b[8 + j] = b1[j]; }

      acc[t] = __builtin_amdgcn_wmma_f32_16x16x32_bf16(
          /*neg_a=*/false, a, /*neg_b=*/false, b,
          /*c_mod=*/(short)0, acc[t], /*reuse_a=*/false, /*reuse_b=*/false);
    }
  }

  // D 16x16 f32: VGPR v, lane(half,sub) -> row M = v + 8*half, col N = sub
#pragma unroll
  for (int t = 0; t < 8; ++t) {
    float* yout = Y + (size_t)rowBase * 128 + t * 16 + sub;
#pragma unroll
    for (int v = 0; v < 8; ++v) yout[(size_t)(v + 8 * half) * 128] = acc[t][v];
  }
}

// --------------------------- edge propagation -----------------------------

// out[i,f] = bias[f] + dinv[i]^2 * Y[i,f]   (self-loop term + bias)
__global__ void prop_init_kernel(const float* __restrict__ Y, const float* __restrict__ dinv,
                                 const float* __restrict__ bias, float* __restrict__ out, int n) {
  size_t idx = (size_t)blockIdx.x * blockDim.x + threadIdx.x;
  if (idx >= (size_t)n * 128) return;
  int i = (int)(idx >> 7);
  int f = (int)(idx & 127);
  float di = dinv[i];
  out[idx] = bias[f] + di * di * Y[idx];
}

// out[col[e], :] += nrm[e] * Y[row[e], :]  (global_atomic_add_f32 scatter)
__global__ void edge_prop_kernel(const int* __restrict__ row, const int* __restrict__ col,
                                 const float* __restrict__ nrm, const float* __restrict__ Y,
                                 float* __restrict__ out, int e) {
  int eo = blockIdx.x * 2 + (int)(threadIdx.x >> 7);
  int f  = (int)(threadIdx.x & 127);
  if (eo >= e) return;
  int r = row[eo], c = col[eo];
  float w = nrm[eo];
  atomicAdd(&out[(size_t)c * 128 + f], w * Y[(size_t)r * 128 + f]);
}

// ------------------------------- launcher ---------------------------------

extern "C" void kernel_launch(void* const* d_in, const int* in_sizes, int n_in,
                              void* d_out, int out_size, void* d_ws, size_t ws_size,
                              hipStream_t stream) {
  (void)n_in; (void)out_size; (void)ws_size;

  const float* x         = (const float*)d_in[0];
  const int*   eidx      = (const int*)d_in[1];
  const float* bn_feat_g = (const float*)d_in[2];
  const float* bn_feat_b = (const float*)d_in[3];
  const float* W_feat    = (const float*)d_in[4];
  const float* b_feat    = (const float*)d_in[5];
  const float* bn_g      = (const float*)d_in[6];
  const float* bn_b      = (const float*)d_in[7];
  const float* Ws        = (const float*)d_in[8];
  const float* bs        = (const float*)d_in[9];

  const int HIN = 256, H = 128;
  const int N = in_sizes[0] / HIN;
  const int E = in_sizes[1] / 2;
  const int* row = eidx;
  const int* col = eidx + E;

  // ---- workspace carve (256B aligned) ----
  char* base = (char*)d_ws;
  size_t off = 0;
  auto carve = [&](size_t bytes) -> void* {
    void* p = base + off;
    off += (bytes + 255) & ~(size_t)255;
    return p;
  };
  float*  deg  = (float*)carve((size_t)N * 4);
  float*  dinv = (float*)carve((size_t)N * 4);
  float*  nrm  = (float*)carve((size_t)E * 4);
  float*  gsum = (float*)carve(256 * 4);
  float*  gsq  = (float*)carve(256 * 4);
  float*  mu   = (float*)carve(256 * 4);
  float*  rsig = (float*)carve(256 * 4);
  bf16_t* Xbf  = (bf16_t*)carve((size_t)N * 256 * 2);
  bf16_t* Wt   = (bf16_t*)carve((size_t)256 * 128 * 2);
  float*  Ybuf = (float*)carve((size_t)N * 128 * 4);
  float*  Xalt = (float*)carve((size_t)N * 128 * 4);
  float*  out  = (float*)d_out;

  // ---- gcn_norm (once, reused by all layers) ----
  init_deg_kernel<<<(N + 255) / 256, 256, 0, stream>>>(deg, N);
  edge_deg_kernel<<<(E + 255) / 256, 256, 0, stream>>>(row, deg, E);
  dinv_kernel<<<(N + 255) / 256, 256, 0, stream>>>(deg, dinv, N);
  enorm_kernel<<<(E + 255) / 256, 256, 0, stream>>>(row, col, dinv, nrm, E);

  // ---- one BN + GCN layer ----
  auto layer = [&](const float* X, int F, const float* g, const float* b,
                   const float* W, const float* bias, float* Xn) {
    bn_zero_kernel<<<1, 256, 0, stream>>>(gsum, gsq, F);
    bn_partial_kernel<<<dim3(F / 32, 64), dim3(32, 8), 0, stream>>>(X, gsum, gsq, N, F);
    bn_finalize_kernel<<<1, 256, 0, stream>>>(gsum, gsq, mu, rsig, N, F);
    size_t tot = (size_t)N * F;
    bn_apply_kernel<<<(unsigned)((tot + 255) / 256), 256, 0, stream>>>(
        X, mu, rsig, g, b, Xbf, tot, F);
    wt_cvt_kernel<<<(128 * F + 255) / 256, 256, 0, stream>>>(W, Wt, F);
    gemm_wmma_kernel<<<N / 16, 32, 0, stream>>>(Xbf, Wt, Ybuf, F);
    prop_init_kernel<<<(unsigned)(((size_t)N * 128 + 255) / 256), 256, 0, stream>>>(
        Ybuf, dinv, bias, Xn, N);
    edge_prop_kernel<<<(E + 1) / 2, 256, 0, stream>>>(row, col, nrm, Ybuf, Xn, E);
  };

  // L1: d_in -> d_out ; L2: d_out -> Xalt ; L3: Xalt -> d_out ; L4: d_out -> d_out
  // (L4 in-place is safe: X is fully consumed by gemm before prop_init rewrites it)
  layer(x,    HIN, bn_feat_g, bn_feat_b, W_feat,          b_feat,     out);
  layer(out,  H,   bn_g + 0 * H, bn_b + 0 * H, Ws + 0 * H * H, bs + 0 * H, Xalt);
  layer(Xalt, H,   bn_g + 1 * H, bn_b + 1 * H, Ws + 1 * H * H, bs + 1 * H, out);
  layer(out,  H,   bn_g + 2 * H, bn_b + 2 * H, Ws + 2 * H * H, bs + 2 * H, out);
}
